// CrossModalAttention_3642132267378
// MI455X (gfx1250) — matrix operand order repaired and checked by reading, not compiled, not run on hardware
//
#include <hip/hip_runtime.h>

// ---------------------------------------------------------------------------
// CrossModalAttention for MI455X (gfx1250, wave32, WMMA bf16 16x16x32)
//   B=4, C=256 channels, N=64*64=4096 pixels
//   Kernel 1: fused concat + 1x1-conv QKV projection  -> bf16 Qt/Kt/V in d_ws
//   Kernel 2: flash attention (online softmax), fp32 accumulate, fp32 out
// ---------------------------------------------------------------------------

typedef __attribute__((ext_vector_type(16))) __bf16 v16bf;
typedef __attribute__((ext_vector_type(8)))  float  v8f;

#define NB    4
#define C_CH  256
#define NPIX  4096

static __device__ __forceinline__ unsigned short f2bf(float x) {
  __bf16 h = (__bf16)x;
  return __builtin_bit_cast(unsigned short, h);
}

union Frag2x16B { uint4 u[2]; v16bf v; };

// Load a 16-element bf16 fragment as two 16-byte vector loads.
static __device__ __forceinline__ v16bf load_frag(const unsigned short* p0,
                                                  const unsigned short* p1) {
  Frag2x16B f;
  f.u[0] = *reinterpret_cast<const uint4*>(p0);
  f.u[1] = *reinterpret_cast<const uint4*>(p1);
  return f.v;
}

static __device__ __forceinline__ float4 ld4(const float* p) {
  return *reinterpret_cast<const float4*>(p);
}

// fp32 -> bf16 fragment (A operand from fp32 weights)
static __device__ __forceinline__ v16bf cvt_frag(float4 a, float4 b, float4 c, float4 d) {
  v16bf v;
  v[0]=(__bf16)a.x;  v[1]=(__bf16)a.y;  v[2]=(__bf16)a.z;  v[3]=(__bf16)a.w;
  v[4]=(__bf16)b.x;  v[5]=(__bf16)b.y;  v[6]=(__bf16)b.z;  v[7]=(__bf16)b.w;
  v[8]=(__bf16)c.x;  v[9]=(__bf16)c.y;  v[10]=(__bf16)c.z; v[11]=(__bf16)c.w;
  v[12]=(__bf16)d.x; v[13]=(__bf16)d.y; v[14]=(__bf16)d.z; v[15]=(__bf16)d.w;
  return v;
}

static __device__ __forceinline__ v8f wmma_bf16(v16bf a, v16bf b, v8f c) {
  // D = A(16x32 bf16) * B(32x16 bf16) + C(16x16 f32)
  return __builtin_amdgcn_wmma_f32_16x16x32_bf16(false, a, false, b,
                                                 (short)0, c, false, false);
}

// Same-wave LDS RAW fence: CDNA5 split counter s_wait_dscnt, plus compiler
// scheduling barriers so the ds stores/loads are not reordered around it.
static __device__ __forceinline__ void lds_sync_wave() {
  __builtin_amdgcn_wave_barrier();
  asm volatile("s_wait_dscnt 0" ::: "memory");
  __builtin_amdgcn_wave_barrier();
}

// ---------------------------------------------------------------------------
// Kernel 1: QKV projection.
// Per wave: one 16(o) x 16(n) tile of q,k,v; K-loop over 256 channels in
// steps of 32. x tile staged through LDS as bf16 [n][c] so the B fragment is
// two contiguous ds_load_b128 per lane.
// Outputs: Qt,Kt = [B][N][C] bf16  (rows contiguous over channels)
//          V     = [B][C][N] bf16  (rows contiguous over pixels)
// ---------------------------------------------------------------------------
__global__ __launch_bounds__(256) void qkv_proj_kernel(
    const float* __restrict__ img, const float* __restrict__ label,
    const float* __restrict__ zin,
    const float* __restrict__ wq, const float* __restrict__ bq,
    const float* __restrict__ wk, const float* __restrict__ bk,
    const float* __restrict__ wv, const float* __restrict__ bv,
    unsigned short* __restrict__ Qt, unsigned short* __restrict__ Kt,
    unsigned short* __restrict__ V)
{
  __shared__ unsigned short xsh[8][16 * 32];   // per-wave [n][c] tile, bf16

  const int lane  = threadIdx.x & 31;
  const int wave  = threadIdx.x >> 5;
  const int gid   = blockIdx.x * 8 + wave;     // 16384 tiles total
  const int b     = gid >> 12;                 // 4096 tiles per batch
  const int rem   = gid & 4095;
  const int otile = (rem >> 8) << 4;           // 16 channel tiles
  const int ntile = (rem & 255) << 4;          // 256 pixel tiles
  const int lo = lane & 15, hi = lane >> 4;

  const v8f vzero = {0.f,0.f,0.f,0.f,0.f,0.f,0.f,0.f};
  v8f accq = vzero, acck = vzero, accv = vzero;
  unsigned short* xs = &xsh[wave][0];

  for (int cs = 0; cs < 8; ++cs) {
    const int c0 = cs * 32;
    // Stage x tile (fused concat) -> LDS [16 n][32 c] bf16
    for (int i = lane; i < 512; i += 32) {
      const int cl = i >> 4, nl = i & 15;
      const int c  = c0 + cl;
      const float* row;
      if (c < 128)      row = img   + ((size_t)b * 128 + c        ) * NPIX;
      else if (c < 192) row = label + ((size_t)b * 64  + (c - 128)) * NPIX;
      else              row = zin   + ((size_t)b * 64  + (c - 192)) * NPIX;
      xs[nl * 32 + cl] = f2bf(row[ntile + nl]);
    }
    lds_sync_wave();

    // B fragment (x): lane column n = lo, k = 8*hi + 0..7 and 16 + 8*hi + 0..7
    const unsigned short* bp = xs + lo * 32 + hi * 8;
    const v16bf xb = load_frag(bp, bp + 16);

    // A fragments (weights, fp32->bf16): row o = otile+lo,
    // runs at k = c0 + 8*hi and k = c0 + 16 + 8*hi
    const size_t woff = (size_t)(otile + lo) * C_CH + c0 + hi * 8;
    const float* rq = wq + woff;
    const float* rk = wk + woff;
    const float* rv = wv + woff;
    const v16bf aq = cvt_frag(ld4(rq), ld4(rq + 4), ld4(rq + 16), ld4(rq + 20));
    const v16bf ak = cvt_frag(ld4(rk), ld4(rk + 4), ld4(rk + 16), ld4(rk + 20));
    const v16bf av = cvt_frag(ld4(rv), ld4(rv + 4), ld4(rv + 16), ld4(rv + 20));

    accq = wmma_bf16(aq, xb, accq);
    acck = wmma_bf16(ak, xb, acck);
    accv = wmma_bf16(av, xb, accv);
    lds_sync_wave();   // drain reads before next iteration overwrites the tile
  }

  // Epilogue: add bias, store bf16.
  const int n  = ntile + lo;          // C-layout: lane column = pixel n
  const int ob = otile + hi * 8;      // rows o = ob..ob+7 live in VGPRs 0..7
  union { unsigned short s[8]; uint4 u; } pq, pk;
  #pragma unroll
  for (int v = 0; v < 8; ++v) {
    pq.s[v] = f2bf(accq[v] + bq[ob + v]);
    pk.s[v] = f2bf(acck[v] + bk[ob + v]);
  }
  unsigned short* qdst = Qt + ((size_t)b * NPIX + n) * C_CH + ob;
  unsigned short* kdst = Kt + ((size_t)b * NPIX + n) * C_CH + ob;
  *reinterpret_cast<uint4*>(qdst) = pq.u;
  *reinterpret_cast<uint4*>(kdst) = pk.u;
  #pragma unroll
  for (int v = 0; v < 8; ++v) {
    V[((size_t)b * C_CH + ob + v) * NPIX + n] = f2bf(accv[v] + bv[ob + v]);
  }
}

// ---------------------------------------------------------------------------
// Kernel 2: flash attention. Each wave owns 16 query rows; loops over keys in
// blocks of 32; online softmax in fp32; P tile re-fragmented through LDS;
// O accumulator = 16 x (16x16 f32) tiles (full 256 channels).
// ---------------------------------------------------------------------------
__global__ __launch_bounds__(256) void flash_attn_kernel(
    const unsigned short* __restrict__ Qt, const unsigned short* __restrict__ Kt,
    const unsigned short* __restrict__ V, float* __restrict__ out)
{
  __shared__ unsigned short psh[8][16 * 32];   // per-wave P tile, bf16

  const int lane = threadIdx.x & 31;
  const int wave = threadIdx.x >> 5;
  const int gid  = blockIdx.x * 8 + wave;      // 1024 query tiles total
  const int b    = gid >> 8;
  const int n0   = (gid & 255) << 4;
  const int lo = lane & 15, hi = lane >> 4;

  const unsigned short* Qb = Qt + (size_t)b * NPIX * C_CH;
  const unsigned short* Kb = Kt + (size_t)b * NPIX * C_CH;
  const unsigned short* Vb = V  + (size_t)b * C_CH * NPIX;
  float*                Ob = out + (size_t)b * C_CH * NPIX;

  // Resident Q fragments (16 queries x full 256 channels).
  v16bf qf[8];
  {
    const unsigned short* qrow = Qb + (size_t)(n0 + lo) * C_CH;
    #pragma unroll
    for (int cs = 0; cs < 8; ++cs) {
      const unsigned short* p = qrow + cs * 32 + hi * 8;
      qf[cs] = load_frag(p, p + 16);
    }
  }

  const v8f vzero = {0.f,0.f,0.f,0.f,0.f,0.f,0.f,0.f};
  float rmax[8], rsum[8];
  #pragma unroll
  for (int v = 0; v < 8; ++v) { rmax[v] = -1.0e30f; rsum[v] = 0.0f; }
  v8f acc[16];
  #pragma unroll
  for (int t = 0; t < 16; ++t) acc[t] = vzero;

  unsigned short* ps = &psh[wave][0];

  for (int kb = 0; kb < NPIX; kb += 32) {
    if (kb + 32 < NPIX)
      __builtin_prefetch(Kb + (size_t)(kb + 32 + lo) * C_CH, 0, 0);

    // S = Q^T K for 32 keys: two 16x16 f32 tiles.
    v8f s0 = vzero, s1 = vzero;
    const unsigned short* k0row = Kb + (size_t)(kb + lo)      * C_CH;
    const unsigned short* k1row = Kb + (size_t)(kb + 16 + lo) * C_CH;
    #pragma unroll
    for (int cs = 0; cs < 8; ++cs) {
      const unsigned short* p0 = k0row + cs * 32 + hi * 8;
      const unsigned short* p1 = k1row + cs * 32 + hi * 8;
      const v16bf kf0 = load_frag(p0, p0 + 16);
      const v16bf kf1 = load_frag(p1, p1 + 16);
      s0 = wmma_bf16(qf[cs], kf0, s0);
      s1 = wmma_bf16(qf[cs], kf1, s1);
    }

    // Online softmax (rows = queries; columns striped over 16-lane halves).
    float scale[8];
    #pragma unroll
    for (int v = 0; v < 8; ++v) {
      float mx = fmaxf(s0[v], s1[v]);
      mx = fmaxf(mx, __shfl_xor(mx, 1, 32));
      mx = fmaxf(mx, __shfl_xor(mx, 2, 32));
      mx = fmaxf(mx, __shfl_xor(mx, 4, 32));
      mx = fmaxf(mx, __shfl_xor(mx, 8, 32));
      const float nm = fmaxf(rmax[v], mx);
      scale[v] = __expf(rmax[v] - nm);
      rmax[v]  = nm;
      const float p0e = __expf(s0[v] - nm);
      const float p1e = __expf(s1[v] - nm);
      float rs = p0e + p1e;
      rs += __shfl_xor(rs, 1, 32);
      rs += __shfl_xor(rs, 2, 32);
      rs += __shfl_xor(rs, 4, 32);
      rs += __shfl_xor(rs, 8, 32);
      rsum[v] = rsum[v] * scale[v] + rs;
      const int row = v + hi * 8;
      ps[row * 32 + lo]      = f2bf(p0e);
      ps[row * 32 + 16 + lo] = f2bf(p1e);
    }

    // Rescale O accumulators (row factor lives per-VGPR, consistent layout).
    #pragma unroll
    for (int t = 0; t < 16; ++t)
      #pragma unroll
      for (int v = 0; v < 8; ++v)
        acc[t][v] *= scale[v];

    lds_sync_wave();
    // Re-fragment P (C-layout -> A-layout) from LDS.
    const unsigned short* prow = ps + lo * 32;
    const v16bf pa = load_frag(prow + hi * 8, prow + 16 + hi * 8);

    // O += P * V  (16 channel tiles, K = 32 keys)
    #pragma unroll
    for (int ct = 0; ct < 16; ++ct) {
      const unsigned short* vp = Vb + (size_t)(ct * 16 + lo) * NPIX + kb + hi * 8;
      const v16bf vf = load_frag(vp, vp + 16);
      acc[ct] = wmma_bf16(pa, vf, acc[ct]);
    }
    lds_sync_wave();
  }

  // Epilogue: normalize and store fp32 output [B][C][N].
  float inv[8];
  #pragma unroll
  for (int v = 0; v < 8; ++v) inv[v] = 1.0f / rsum[v];
  #pragma unroll
  for (int ct = 0; ct < 16; ++ct) {
    const int c = ct * 16 + lo;
    float* op = Ob + (size_t)c * NPIX + n0 + hi * 8;
    float4 o0, o1;
    o0.x = acc[ct][0] * inv[0]; o0.y = acc[ct][1] * inv[1];
    o0.z = acc[ct][2] * inv[2]; o0.w = acc[ct][3] * inv[3];
    o1.x = acc[ct][4] * inv[4]; o1.y = acc[ct][5] * inv[5];
    o1.z = acc[ct][6] * inv[6]; o1.w = acc[ct][7] * inv[7];
    *reinterpret_cast<float4*>(op)     = o0;
    *reinterpret_cast<float4*>(op + 4) = o1;
  }
}

// ---------------------------------------------------------------------------
extern "C" void kernel_launch(void* const* d_in, const int* in_sizes, int n_in,
                              void* d_out, int out_size, void* d_ws, size_t ws_size,
                              hipStream_t stream) {
  const float* img   = (const float*)d_in[0];
  const float* label = (const float*)d_in[1];
  const float* zin   = (const float*)d_in[2];
  const float* wq    = (const float*)d_in[3];
  const float* bq    = (const float*)d_in[4];
  const float* wk    = (const float*)d_in[5];
  const float* bk    = (const float*)d_in[6];
  const float* wv    = (const float*)d_in[7];
  const float* bv    = (const float*)d_in[8];

  const size_t NC = (size_t)NB * NPIX * C_CH;   // elements per tensor
  unsigned short* Qt = (unsigned short*)d_ws;
  unsigned short* Kt = Qt + NC;
  unsigned short* V  = Kt + NC;                 // total 3*NC*2 = 24 MB

  // 16384 wave-tiles, 8 waves per 256-thread block
  qkv_proj_kernel<<<2048, 256, 0, stream>>>(img, label, zin,
                                            wq, bq, wk, bk, wv, bv,
                                            Qt, Kt, V);
  // 1024 query wave-tiles, 8 waves per block
  flash_attn_kernel<<<128, 256, 0, stream>>>(Qt, Kt, V, (float*)d_out);
}
